// MultiHeadAttention_67783173865634
// MI455X (gfx1250) — compile-verified, matrix operand order checked by reading, not run on hardware
//
#include <hip/hip_runtime.h>
#include <hip/hip_bf16.h>

// ---------------------------------------------------------------------------
// MI455X (gfx1250) multi-head causal attention, bf16 WMMA pipeline, f32 accum.
// B=4, S=1024, E=1024, H=16, D=64.
// ---------------------------------------------------------------------------

typedef __bf16 v16bf __attribute__((ext_vector_type(16)));
typedef __bf16 v8bf  __attribute__((ext_vector_type(8)));
typedef float  v8f   __attribute__((ext_vector_type(8)));
typedef float  v4f   __attribute__((ext_vector_type(4)));

#define BB 4
#define SS 1024
#define EE 1024
#define HH 16
#define DD 64

__device__ __forceinline__ v8f wmma_bf16(v16bf a, v16bf b, v8f c) {
  // D = A(16x32 bf16) * B(32x16 bf16) + C(16x16 f32)
  return __builtin_amdgcn_wmma_f32_16x16x32_bf16(
      /*neg_a=*/false, a, /*neg_b=*/false, b,
      /*c_mod=*/(short)0, c, /*reuse_a=*/false, /*reuse_b=*/false);
}

// A/B fragment loader from row-major bf16 storage.
// ISA 7.12.2 16-bit 16x32 layout: lane L<16 -> row L, K = kb..kb+7 then kb+16..kb+23
// with kb = (lane>>4)*8. B fragments use identical per-lane pattern with "row" = column.
__device__ __forceinline__ v16bf ld_frag_bf16(const __bf16* base, int stride) {
  const int lane = threadIdx.x & 31;
  const int r    = lane & 15;
  const int kb   = (lane >> 4) << 3;
  const __bf16* p = base + (size_t)r * stride + kb;
  v8bf lo = *(const v8bf*)(p);        // K = kb .. kb+7      (16B aligned)
  v8bf hi = *(const v8bf*)(p + 16);   // K = kb+16 .. kb+23  (16B aligned)
  v16bf o;
#pragma unroll
  for (int j = 0; j < 8; ++j) { o[j] = lo[j]; o[j + 8] = hi[j]; }
  return o;
}

// Same fragment pattern, source f32, converted to bf16 on the fly.
__device__ __forceinline__ v16bf ld_frag_f32(const float* base, int stride) {
  const int lane = threadIdx.x & 31;
  const int r    = lane & 15;
  const int kb   = (lane >> 4) << 3;
  const float* p = base + (size_t)r * stride + kb;
  v4f a0 = *(const v4f*)(p);
  v4f a1 = *(const v4f*)(p + 4);
  v4f b0 = *(const v4f*)(p + 16);
  v4f b1 = *(const v4f*)(p + 20);
  v16bf o;
#pragma unroll
  for (int j = 0; j < 4; ++j) {
    o[j]      = (__bf16)a0[j];
    o[j + 4]  = (__bf16)a1[j];
    o[j + 8]  = (__bf16)b0[j];
    o[j + 12] = (__bf16)b1[j];
  }
  return o;
}

// Prefetch helper (probe-confirmed: lowers to global_prefetch_b8 on gfx1250).
__device__ __forceinline__ void pf(const void* p) { __builtin_prefetch(p, 0, 3); }

// ---------------------------------------------------------------------------
// Kernel 0a: Wq/Wk/Wv (H,D,D) f32 -> transposed bf16 WqT[h][n][k] = Wq[h][k][n]
// ---------------------------------------------------------------------------
__global__ __launch_bounds__(256) void cvt_qkv_kernel(
    const float* __restrict__ Wq, const float* __restrict__ Wk,
    const float* __restrict__ Wv, __bf16* __restrict__ WqT,
    __bf16* __restrict__ WkT, __bf16* __restrict__ WvT) {
  int i = blockIdx.x * 256 + threadIdx.x;        // < H*D*D = 65536
  int h = i >> 12, rem = i & 4095;
  int n = rem >> 6, k = rem & 63;
  int src = h * 4096 + k * 64 + n;
  WqT[i] = (__bf16)Wq[src];
  WkT[i] = (__bf16)Wk[src];
  WvT[i] = (__bf16)Wv[src];
}

// Kernel 0b: W_O (E,E) f32 -> WoT[j][k] = W_O[k][j] bf16
__global__ __launch_bounds__(256) void cvt_wo_kernel(
    const float* __restrict__ Wo, __bf16* __restrict__ WoT) {
  int i = blockIdx.x * 256 + threadIdx.x;        // < E*E
  int j = i >> 10, k = i & 1023;
  WoT[i] = (__bf16)Wo[k * EE + j];
}

// ---------------------------------------------------------------------------
// Kernel 1: per-head Q/K/V projection. One wave per (b,h, 16-row tile).
// Q scaled by 1/sqrt(D)=0.125 at store. V stored transposed: Vt[bh][d][s].
// ---------------------------------------------------------------------------
__global__ __launch_bounds__(256) void qkv_kernel(
    const float* __restrict__ x, const __bf16* __restrict__ WqT,
    const __bf16* __restrict__ WkT, const __bf16* __restrict__ WvT,
    __bf16* __restrict__ Q, __bf16* __restrict__ K, __bf16* __restrict__ Vt) {
  const int lane = threadIdx.x & 31;
  const int w    = blockIdx.x * 8 + (threadIdx.x >> 5);   // 0..4095
  const int qt = w & 63, bh = w >> 6;
  const int h = bh & 15, b = bh >> 4;
  const int q0 = qt * 16;
  const int ls = lane & 15, hi = lane >> 4;

  const float* xb = x + ((size_t)b * SS + q0) * EE + h * DD;

  v8f aq[4], ak[4], av[4];
#pragma unroll
  for (int t = 0; t < 4; ++t)
#pragma unroll
    for (int e = 0; e < 8; ++e) { aq[t][e] = 0.f; ak[t][e] = 0.f; av[t][e] = 0.f; }

#pragma unroll
  for (int ks = 0; ks < 2; ++ks) {                        // K-dim: D=64 -> 2x32
    v16bf a = ld_frag_f32(xb + ks * 32, EE);
#pragma unroll
    for (int nt = 0; nt < 4; ++nt) {
      const int off = h * 4096 + nt * 16 * 64 + ks * 32;  // [h][n][k], stride 64
      aq[nt] = wmma_bf16(a, ld_frag_bf16(WqT + off, 64), aq[nt]);
      ak[nt] = wmma_bf16(a, ld_frag_bf16(WkT + off, 64), ak[nt]);
      av[nt] = wmma_bf16(a, ld_frag_bf16(WvT + off, 64), av[nt]);
    }
  }

  const size_t qkbase = ((size_t)bh * SS + q0) * DD;      // Q/K: [bh][s][d]
  const size_t vtbase = (size_t)bh * DD * SS;             // Vt:  [bh][d][s]
#pragma unroll
  for (int nt = 0; nt < 4; ++nt) {
#pragma unroll
    for (int r = 0; r < 8; ++r) {
      const int row = r + hi * 8;                         // seq within tile
      const int d   = nt * 16 + ls;
      Q[qkbase + (size_t)row * DD + d] = (__bf16)(aq[nt][r] * 0.125f);
      K[qkbase + (size_t)row * DD + d] = (__bf16)ak[nt][r];
      Vt[vtbase + (size_t)d * SS + (q0 + row)] = (__bf16)av[nt][r];
    }
  }
}

// ---------------------------------------------------------------------------
// Kernel 2: causal flash attention, online softmax. One wave per (b,h,q-tile).
// 32 keys per iteration: 4 WMMA for scores + 4 WMMA for P*V.
// Query tiles remapped so adjacent waves pair tile t with tile 63-t: every
// pair does a uniform 33 key-blocks -> block-level load balance under the
// triangular causal work distribution.
// Writes catT[b][emb][seq] (absorbs the reference's S==E transpose "bug").
// ---------------------------------------------------------------------------
__global__ __launch_bounds__(256) void attn_kernel(
    const __bf16* __restrict__ Q, const __bf16* __restrict__ K,
    const __bf16* __restrict__ Vt, __bf16* __restrict__ catT) {
  __shared__ __align__(16) __bf16 plds[8][16 * 32];       // per-wave P tile
  const int lane = threadIdx.x & 31;
  const int wv   = threadIdx.x >> 5;
  const int w    = blockIdx.x * 8 + wv;                   // 0..4095
  const int t_   = w & 63;
  const int qt   = (t_ & 1) ? (63 - (t_ >> 1)) : (t_ >> 1);  // balanced pairing
  const int bh   = w >> 6;
  const int q0 = qt * 16;
  const int ls = lane & 15, hi = lane >> 4;

  const __bf16* Qb = Q + ((size_t)bh * SS + q0) * DD;
  const __bf16* Kb = K + (size_t)bh * SS * DD;
  const __bf16* Vb = Vt + (size_t)bh * DD * SS;

  v8f o[4];
  float m[8], l[8];
#pragma unroll
  for (int t = 0; t < 4; ++t)
#pragma unroll
    for (int e = 0; e < 8; ++e) o[t][e] = 0.f;
#pragma unroll
  for (int r = 0; r < 8; ++r) { m[r] = -1e30f; l[r] = 0.f; }

  const int nkb = (q0 + 47) >> 5;                         // causal key blocks of 32
  for (int kb = 0; kb < nkb; ++kb) {
    const int key0 = kb * 32;

    // Prefetch next key block's K rows and V rows while this block computes.
    if (kb + 1 < nkb) {
      const int nk = key0 + 32;
      pf(Kb + (size_t)(nk + (lane & 31)) * DD);           // next 32 K rows
      pf(Vb + (size_t)(lane & 31) * SS + nk);             // next V columns, d=0..31
      pf(Vb + (size_t)(32 + (lane & 31)) * SS + nk);      // d=32..63
    }

    v8f s0, s1;
#pragma unroll
    for (int e = 0; e < 8; ++e) { s0[e] = 0.f; s1[e] = 0.f; }

#pragma unroll
    for (int ks = 0; ks < 2; ++ks) {                      // contract over D=64
      v16bf a  = ld_frag_bf16(Qb + ks * 32, DD);
      v16bf b0 = ld_frag_bf16(Kb + (size_t)key0 * DD + ks * 32, DD);
      v16bf b1 = ld_frag_bf16(Kb + (size_t)(key0 + 16) * DD + ks * 32, DD);
      s0 = wmma_bf16(a, b0, s0);
      s1 = wmma_bf16(a, b1, s1);
    }

    // online softmax over this 32-key block (C-frag row = r + hi*8, col = ls)
#pragma unroll
    for (int r = 0; r < 8; ++r) {
      const int row = q0 + r + hi * 8;
      const int c0 = key0 + ls, c1 = c0 + 16;
      float v0 = (c0 <= row) ? s0[r] : -1e30f;
      float v1 = (c1 <= row) ? s1[r] : -1e30f;
      float mt = fmaxf(v0, v1);
#pragma unroll
      for (int d = 1; d < 16; d <<= 1) mt = fmaxf(mt, __shfl_xor(mt, d, 32));
      const float mn    = fmaxf(m[r], mt);
      const float alpha = __expf(m[r] - mn);
      const float p0 = __expf(v0 - mn);
      const float p1 = __expf(v1 - mn);
      float rs = p0 + p1;
#pragma unroll
      for (int d = 1; d < 16; d <<= 1) rs += __shfl_xor(rs, d, 32);
      l[r] = l[r] * alpha + rs;
      m[r] = mn;
#pragma unroll
      for (int t = 0; t < 4; ++t) o[t][r] *= alpha;
      const int prow = r + hi * 8;
      plds[wv][prow * 32 + ls]      = (__bf16)p0;
      plds[wv][prow * 32 + ls + 16] = (__bf16)p1;
    }
    asm volatile("s_wait_dscnt 0" ::: "memory");          // same-wave LDS RAW

    v16bf pa = ld_frag_bf16(&plds[wv][0], 32);            // P as A-fragment
#pragma unroll
    for (int t = 0; t < 4; ++t) {                         // P(16x32) * V(32x16)
      v16bf vb = ld_frag_bf16(Vb + (size_t)(t * 16) * SS + key0, SS);
      o[t] = wmma_bf16(pa, vb, o[t]);
    }
  }

  const int b = bh >> 4, h = bh & 15;
#pragma unroll
  for (int t = 0; t < 4; ++t) {
#pragma unroll
    for (int r = 0; r < 8; ++r) {
      const int srow = q0 + r + hi * 8;
      const int emb  = h * DD + t * 16 + ls;
      const float val = o[t][r] / l[r];
      catT[((size_t)b * EE + emb) * SS + srow] = (__bf16)val;
    }
  }
}

// ---------------------------------------------------------------------------
// Kernel 3: out[b,i,j] = sum_k cat[b,k,i] * W_O[k,j]  ==  catT[b] @ W_O.
// One wave = 16x64 output tile (4 N-tiles share the A fragment), 32 K-steps.
// ---------------------------------------------------------------------------
__global__ __launch_bounds__(256) void out_kernel(
    const __bf16* __restrict__ catT, const __bf16* __restrict__ WoT,
    float* __restrict__ out) {
  const int lane = threadIdx.x & 31;
  const int w    = blockIdx.x * 8 + (threadIdx.x >> 5);   // 0..4095
  const int b   = w >> 10;
  const int rem = w & 1023;
  const int it  = rem >> 4;                               // 16-row tile of i
  const int jg  = rem & 15;                               // 64-col group of j
  const int ls = lane & 15, hi = lane >> 4;

  const __bf16* ab = catT + ((size_t)b * EE + it * 16) * SS;
  const __bf16* bb0 = WoT + (size_t)(jg * 64) * EE;

  v8f acc[4];
#pragma unroll
  for (int t = 0; t < 4; ++t)
#pragma unroll
    for (int e = 0; e < 8; ++e) acc[t][e] = 0.f;

  for (int ks = 0; ks < SS / 32; ++ks) {
    // Prefetch next K-step's A rows (64B lines cover both 32B runs).
    if (ks + 1 < SS / 32)
      pf(ab + (size_t)(lane & 15) * SS + (ks + 1) * 32);

    v16bf a = ld_frag_bf16(ab + ks * 32, SS);
#pragma unroll
    for (int jt = 0; jt < 4; ++jt) {
      v16bf bb = ld_frag_bf16(bb0 + (size_t)(jt * 16) * EE + ks * 32, EE);
      acc[jt] = wmma_bf16(a, bb, acc[jt]);
    }
  }

#pragma unroll
  for (int jt = 0; jt < 4; ++jt) {
#pragma unroll
    for (int r = 0; r < 8; ++r) {
      const int i = it * 16 + r + hi * 8;
      const int j = jg * 64 + jt * 16 + ls;
      out[((size_t)b * SS + i) * EE + j] = acc[jt][r];
    }
  }
}

// ---------------------------------------------------------------------------
extern "C" void kernel_launch(void* const* d_in, const int* in_sizes, int n_in,
                              void* d_out, int out_size, void* d_ws, size_t ws_size,
                              hipStream_t stream) {
  const float* x  = (const float*)d_in[0];   // (B,S,E)
  const float* Wq = (const float*)d_in[1];   // (H,D,D)
  const float* Wk = (const float*)d_in[2];
  const float* Wv = (const float*)d_in[3];
  const float* Wo = (const float*)d_in[4];   // (E,E)
  float* out = (float*)d_out;                // (B,S,E)

  char* ws = (char*)d_ws;
  __bf16* Q    = (__bf16*)(ws + (size_t)0);            // 8 MB  [bh][s][d], pre-scaled
  __bf16* K    = (__bf16*)(ws + ((size_t)8  << 20));   // 8 MB  [bh][s][d]
  __bf16* Vt   = (__bf16*)(ws + ((size_t)16 << 20));   // 8 MB  [bh][d][s]
  __bf16* catT = (__bf16*)(ws + ((size_t)24 << 20));   // 8 MB  [b][emb][s]
  __bf16* WqT  = (__bf16*)(ws + ((size_t)32 << 20));   // 128 KB each
  __bf16* WkT  = WqT + 65536;
  __bf16* WvT  = WkT + 65536;
  __bf16* WoT  = WvT + 65536;                          // 2 MB [j][k]

  cvt_qkv_kernel<<<256, 256, 0, stream>>>(Wq, Wk, Wv, WqT, WkT, WvT);
  cvt_wo_kernel<<<4096, 256, 0, stream>>>(Wo, WoT);
  qkv_kernel<<<512, 256, 0, stream>>>(x, WqT, WkT, WvT, Q, K, Vt);
  attn_kernel<<<512, 256, 0, stream>>>(Q, K, Vt, catT);
  out_kernel<<<512, 256, 0, stream>>>(catT, WoT, out);
}
